// CrossBandWindowAttention_30356828848467
// MI455X (gfx1250) — compile-verified, matrix operand order checked by reading, not run on hardware
//
#include <hip/hip_runtime.h>
#include <hip/hip_fp16.h>

typedef __attribute__((ext_vector_type(16))) _Float16 v16h;
typedef __attribute__((ext_vector_type(8)))  float    v8f;
typedef __attribute__((ext_vector_type(4)))  float    v4f;
typedef __attribute__((ext_vector_type(4)))  unsigned int v4u;
typedef __attribute__((ext_vector_type(8)))  unsigned int v8u;

#define DIMC   384
#define HEADS  12
#define HD     32
#define NTOK   64
#define BWIN   4096
#define NWIN   1024
#define TABLE  225

union F16x16 {
    v16h h;
    struct { v4u lo, hi; } p;
};

// ---------------------------------------------------------------------------
// TDM: async 2D f16 tile load Global -> LDS via Tensor Data Mover.
// D# built per CDNA5 ISA ch.8: group0 (count/lds_addr/global_addr/type),
// group1 (data_size, pad config, dims, strides). 2-group form (<=2D tensor).
// Pad: every 16 DWORDs (one 64B row) insert 4 DWORDs -> 80B LDS row stride.
// ---------------------------------------------------------------------------
__device__ __forceinline__ void tdm_load_tile_f16(
    unsigned lds_off, const _Float16* gptr,
    unsigned tile_w, unsigned tile_h, unsigned row_stride)
{
    const unsigned long long ga = (unsigned long long)(size_t)gptr;
    v4u g0;
    g0.x = 0x1u;                                   // count=1, user descriptor
    g0.y = lds_off;                                // lds_addr (bytes)
    g0.z = (unsigned)(ga & 0xFFFFFFFFu);           // global_addr[31:0]
    g0.w = (unsigned)((ga >> 32) & 0x01FFFFFFu)    // global_addr[56:32]
         | (2u << 30);                             // type=2 ("image")
    v8u g1;
    g1.s0 = (1u << 16)                             // data_size = 2 bytes
          | (1u << 20)                             // pad_enable
          | (3u << 22)                             // pad_interval: 16 DWORDs
          | (3u << 25);                            // pad_amount:   4 DWORDs
    g1.s1 = (row_stride & 0xFFFFu) << 16;          // tensor_dim0[15:0]
    g1.s2 = ((row_stride >> 16) & 0xFFFFu)         // tensor_dim0[31:16]
          | (0x8000u << 16);                       // tensor_dim1[15:0] (large)
    g1.s3 = (tile_w & 0xFFFFu) << 16;              // tile_dim0 (dim1 hi16 = 0)
    g1.s4 = tile_h & 0xFFFFu;                      // tile_dim1 (tile_dim2 = 0)
    g1.s5 = row_stride;                            // tensor_dim0_stride[31:0]
    g1.s6 = 0u;                                    // stride0 hi / stride1 lo
    g1.s7 = 0u;
    asm volatile("tensor_load_to_lds %0, %1" :: "s"(g0), "s"(g1) : "memory");
}

// ---------------------------------------------------------------------------
// Prep: f32 weights -> f16; gather relative-position bias table.
// ---------------------------------------------------------------------------
__global__ __launch_bounds__(256) void prep_kernel(
    const float* __restrict__ wq, const float* __restrict__ wkv,
    const float* __restrict__ wproj, const int* __restrict__ rpi,
    const float* __restrict__ bias_table,
    _Float16* __restrict__ wq16, _Float16* __restrict__ wkv16,
    _Float16* __restrict__ wproj16, float* __restrict__ rpb)
{
    int i = blockIdx.x * 256 + threadIdx.x;
    if (i < DIMC * DIMC)     wq16[i]    = (_Float16)wq[i];
    if (i < 2 * DIMC * DIMC) wkv16[i]   = (_Float16)wkv[i];
    if (i < DIMC * DIMC)     wproj16[i] = (_Float16)wproj[i];
    if (i < HEADS * NTOK * NTOK) {
        int h  = i / (NTOK * NTOK);
        int qk = i % (NTOK * NTOK);
        rpb[i] = bias_table[rpi[qk] * HEADS + h];
    }
}

// ---------------------------------------------------------------------------
// WMMA GEMM: C[M x NOUT] = A[M x 384] * W^T  (W stored [NOUT][384] f16)
// MODE 0: A=f32 x,        out -> q16  [b][h][n][d] f16, scaled by hd^-0.5
// MODE 1: A=f32 cross_x,  out -> k16  [b][h][n][d] f16  (cols <384)
//                               v16t [b][h][d][n] f16  (cols >=384)
// MODE 2: A=f16 att,      out -> f32 [b][n][c] (+bproj)
// Block: 256 threads = 8 waves; block tile 128M x 64N; wave tile 32x32.
// B tile staged by the Tensor Data Mover; A tile by VALU f32->f16 convert.
// ---------------------------------------------------------------------------
template <int MODE>
__global__ __launch_bounds__(256) void gemm_wmma(
    const void* __restrict__ Aglob,
    const _Float16* __restrict__ Wf16,
    const float* __restrict__ bias,
    void* __restrict__ Out0,
    _Float16* __restrict__ Out1)
{
    constexpr int K = DIMC;
    __shared__ _Float16 Atile[128][40];   // 80B row stride: 16B aligned, padded
    __shared__ _Float16 Btile[64][40];    // TDM-landed (pad -> same 80B stride)

    const int tid  = threadIdx.x;
    const int wave = tid >> 5;
    const int lane = tid & 31;
    const int l16  = lane & 15;
    const int lh   = lane >> 4;           // lane half selects K sub-range
    const int m0   = blockIdx.y * 128;
    const int n0   = blockIdx.x * 64;
    const int wm   = (wave & 3) * 32;
    const int wn   = (wave >> 2) * 32;

    const int srow  = tid >> 1;           // staging: row 0..127
    const int shalf = tid & 1;            // 16-element half of the K-slice

    const unsigned bt_lds = (unsigned)(size_t)(const void*)&Btile[0][0];

    v8f acc[2][2] = {};

    for (int kb = 0; kb < K; kb += 32) {
        // ---- kick the TDM for the B tile (64 rows x 32 f16, stride 384) ----
        if (wave == 0)
            tdm_load_tile_f16(bt_lds, Wf16 + (size_t)n0 * K + kb, 32, 64, K);

        // ---- stage A tile (f32->f16 convert, or straight f16 copy) ----
        if (MODE != 2) {
            const float* ap = (const float*)Aglob +
                              (size_t)(m0 + srow) * K + kb + shalf * 16;
            _Float16 tmp[16];
#pragma unroll
            for (int i = 0; i < 16; i += 4) {
                v4f f = *(const v4f*)(ap + i);
                tmp[i + 0] = (_Float16)f.x; tmp[i + 1] = (_Float16)f.y;
                tmp[i + 2] = (_Float16)f.z; tmp[i + 3] = (_Float16)f.w;
            }
            *(v4u*)&Atile[srow][shalf * 16 + 0] = *(v4u*)&tmp[0];
            *(v4u*)&Atile[srow][shalf * 16 + 8] = *(v4u*)&tmp[8];
            if (kb + 32 < K) __builtin_prefetch(ap + 32, 0, 0);
        } else {
            const _Float16* ap = (const _Float16*)Aglob +
                                 (size_t)(m0 + srow) * K + kb + shalf * 16;
            *(v4u*)&Atile[srow][shalf * 16 + 0] = *(const v4u*)(ap + 0);
            *(v4u*)&Atile[srow][shalf * 16 + 8] = *(const v4u*)(ap + 8);
        }

        // issuing wave drains TENSORcnt, then the barrier publishes both tiles
        if (wave == 0) __builtin_amdgcn_s_wait_tensorcnt(0);
        __syncthreads();

        // ---- A fragments from LDS (lh picks K 0-7/16-23 vs 8-15/24-31) ----
        v16h afrag[2];
#pragma unroll
        for (int mi = 0; mi < 2; mi++) {
            const _Float16* p = &Atile[wm + mi * 16 + l16][lh * 8];
            F16x16 t; t.p.lo = *(const v4u*)p; t.p.hi = *(const v4u*)(p + 16);
            afrag[mi] = t.h;
        }
        // ---- B fragments from the TDM-landed LDS tile ----
        v16h bfrag[2];
#pragma unroll
        for (int ni = 0; ni < 2; ni++) {
            const _Float16* p = &Btile[wn + ni * 16 + l16][lh * 8];
            F16x16 t; t.p.lo = *(const v4u*)p; t.p.hi = *(const v4u*)(p + 16);
            bfrag[ni] = t.h;
        }
#pragma unroll
        for (int mi = 0; mi < 2; mi++)
#pragma unroll
            for (int ni = 0; ni < 2; ni++)
                acc[mi][ni] = __builtin_amdgcn_wmma_f32_16x16x32_f16(
                    false, afrag[mi], false, bfrag[ni],
                    (short)0, acc[mi][ni], false, false);
        __syncthreads();
    }

    // ---- epilogue: bias, scale, scatter to mode-specific layout ----
    const float qscale = 0.17677669529663689f;   // hd^-0.5
#pragma unroll
    for (int mi = 0; mi < 2; mi++)
#pragma unroll
        for (int ni = 0; ni < 2; ni++)
#pragma unroll
            for (int j = 0; j < 8; j++) {
                const int gm = m0 + wm + mi * 16 + lh * 8 + j;   // token row
                const int gc = n0 + wn + ni * 16 + l16;          // out col
                float v = acc[mi][ni][j] + bias[gc];
                const int b = gm >> 6, n = gm & 63;
                if (MODE == 0) {
                    v *= qscale;
                    const int h = gc >> 5, d = gc & 31;
                    ((_Float16*)Out0)[(((size_t)b * HEADS + h) * NTOK + n) * HD + d] =
                        (_Float16)v;
                } else if (MODE == 1) {
                    if (gc < DIMC) {                 // K part: [b][h][n][d]
                        const int h = gc >> 5, d = gc & 31;
                        ((_Float16*)Out0)[(((size_t)b * HEADS + h) * NTOK + n) * HD + d] =
                            (_Float16)v;
                    } else {                         // V part transposed: [b][h][d][n]
                        const int cv = gc - DIMC;
                        const int h = cv >> 5, d = cv & 31;
                        Out1[(((size_t)b * HEADS + h) * HD + d) * NTOK + n] =
                            (_Float16)v;
                    }
                } else {                              // MODE 2: final f32 output
                    ((float*)Out0)[(size_t)gm * DIMC + gc] = v;
                }
            }
}

// ---------------------------------------------------------------------------
// Attention: one block per (b,h). 128 threads = 4 waves; wave w owns query
// rows [16w, 16w+16). hd=32 -> QK^T is a single 16x16x32 WMMA per tile.
// ---------------------------------------------------------------------------
__global__ __launch_bounds__(128) void attn_kernel(
    const _Float16* __restrict__ q16,   // [b][h][64][32]
    const _Float16* __restrict__ k16,   // [b][h][64][32]
    const _Float16* __restrict__ v16t,  // [b][h][32][64]
    const float* __restrict__ rpb,      // [h][64][64]
    const float* __restrict__ mask,     // [NWIN][64][64]
    _Float16* __restrict__ att16)       // [b][64][384]
{
    __shared__ _Float16 P[64][72];      // 144B row stride, 16B aligned

    const int bh   = blockIdx.x;
    const int b    = bh / HEADS;
    const int h    = bh % HEADS;
    const int wave = threadIdx.x >> 5;
    const int lane = threadIdx.x & 31;
    const int l16  = lane & 15;
    const int lh   = lane >> 4;

    const _Float16* qb = q16  + (size_t)bh * NTOK * HD;
    const _Float16* kb = k16  + (size_t)bh * NTOK * HD;
    const _Float16* vb = v16t + (size_t)bh * HD * NTOK;

    // Q fragment for this wave's 16 rows (K=32, one fragment)
    v16h qf;
    {
        const _Float16* p = qb + (size_t)(wave * 16 + l16) * HD + lh * 8;
        F16x16 t; t.p.lo = *(const v4u*)p; t.p.hi = *(const v4u*)(p + 16);
        qf = t.h;
    }

    // scores = q @ k^T : 4 column tiles, one WMMA each
    v8f sc[4];
#pragma unroll
    for (int c = 0; c < 4; c++) {
        const _Float16* p = kb + (size_t)(c * 16 + l16) * HD + lh * 8;
        F16x16 t; t.p.lo = *(const v4u*)p; t.p.hi = *(const v4u*)(p + 16);
        v8f z = {};
        sc[c] = __builtin_amdgcn_wmma_f32_16x16x32_f16(
            false, qf, false, t.h, (short)0, z, false, false);
    }

    // + relative position bias + shift mask
    const int w = b & (NWIN - 1);
    const float* rp = rpb  + (size_t)h * NTOK * NTOK;
    const float* mp = mask + (size_t)w * NTOK * NTOK;
#pragma unroll
    for (int c = 0; c < 4; c++)
#pragma unroll
        for (int j = 0; j < 8; j++) {
            const int r   = wave * 16 + lh * 8 + j;
            const int col = c * 16 + l16;
            sc[c][j] += rp[r * NTOK + col] + mp[r * NTOK + col];
        }

    // row softmax: 4 in-lane values + 16-lane group reduce (C layout keeps a
    // row's N=0..15 inside one 16-lane half, so xor masks 1..8 suffice)
#pragma unroll
    for (int j = 0; j < 8; j++) {
        float mx = fmaxf(fmaxf(sc[0][j], sc[1][j]), fmaxf(sc[2][j], sc[3][j]));
#pragma unroll
        for (int m = 1; m <= 8; m <<= 1) mx = fmaxf(mx, __shfl_xor(mx, m, 32));
        float e0 = __expf(sc[0][j] - mx), e1 = __expf(sc[1][j] - mx);
        float e2 = __expf(sc[2][j] - mx), e3 = __expf(sc[3][j] - mx);
        float s = e0 + e1 + e2 + e3;
#pragma unroll
        for (int m = 1; m <= 8; m <<= 1) s += __shfl_xor(s, m, 32);
        const float inv = 1.0f / s;
        sc[0][j] = e0 * inv; sc[1][j] = e1 * inv;
        sc[2][j] = e2 * inv; sc[3][j] = e3 * inv;
    }

    // P -> LDS (f16) so it can be re-read in the WMMA A-fragment layout
#pragma unroll
    for (int c = 0; c < 4; c++)
#pragma unroll
        for (int j = 0; j < 8; j++)
            P[wave * 16 + lh * 8 + j][c * 16 + l16] = (_Float16)sc[c][j];
    __syncthreads();

    // out = P @ V : K=64 in two steps of 32; 2 column tiles (d = 0..31)
    v8f oacc[2] = {};
#pragma unroll
    for (int s = 0; s < 2; s++) {
        v16h pf;
        {
            const _Float16* p = &P[wave * 16 + l16][s * 32 + lh * 8];
            F16x16 t; t.p.lo = *(const v4u*)p; t.p.hi = *(const v4u*)(p + 16);
            pf = t.h;
        }
#pragma unroll
        for (int ni = 0; ni < 2; ni++) {
            const _Float16* p = vb + (size_t)(ni * 16 + l16) * NTOK + s * 32 + lh * 8;
            F16x16 t; t.p.lo = *(const v4u*)p; t.p.hi = *(const v4u*)(p + 16);
            oacc[ni] = __builtin_amdgcn_wmma_f32_16x16x32_f16(
                false, pf, false, t.h, (short)0, oacc[ni], false, false);
        }
    }

    // store to [b][n][c] f16 for the projection GEMM
    _Float16* ob = att16 + (size_t)b * NTOK * DIMC + h * HD;
#pragma unroll
    for (int ni = 0; ni < 2; ni++)
#pragma unroll
        for (int j = 0; j < 8; j++) {
            const int r = wave * 16 + lh * 8 + j;
            ob[(size_t)r * DIMC + ni * 16 + l16] = (_Float16)oacc[ni][j];
        }
}

// ---------------------------------------------------------------------------
extern "C" void kernel_launch(void* const* d_in, const int* in_sizes, int n_in,
                              void* d_out, int out_size, void* d_ws, size_t ws_size,
                              hipStream_t stream)
{
    const float* x          = (const float*)d_in[0];
    const float* cross_x    = (const float*)d_in[1];
    const int*   rpi        = (const int*)  d_in[2];
    const float* mask       = (const float*)d_in[3];
    const float* wq         = (const float*)d_in[4];
    const float* bq         = (const float*)d_in[5];
    const float* wkv        = (const float*)d_in[6];
    const float* bkv        = (const float*)d_in[7];
    const float* bias_table = (const float*)d_in[8];
    const float* wproj      = (const float*)d_in[9];
    const float* bproj      = (const float*)d_in[10];
    float* out = (float*)d_out;

    // workspace carve-up (256B aligned)
    char* base = (char*)d_ws;
    size_t off = 0;
    auto carve = [&](size_t bytes) {
        char* p = base + off;
        off = (off + bytes + 255) & ~(size_t)255;
        return p;
    };
    const size_t actBytes = (size_t)BWIN * NTOK * DIMC * sizeof(_Float16);
    _Float16* q16     = (_Float16*)carve(actBytes);
    _Float16* k16     = (_Float16*)carve(actBytes);
    _Float16* v16t    = (_Float16*)carve(actBytes);
    _Float16* att16   = (_Float16*)carve(actBytes);
    _Float16* wq16    = (_Float16*)carve((size_t)DIMC * DIMC * 2);
    _Float16* wkv16   = (_Float16*)carve((size_t)2 * DIMC * DIMC * 2);
    _Float16* wproj16 = (_Float16*)carve((size_t)DIMC * DIMC * 2);
    float*    rpb     = (float*)   carve((size_t)HEADS * NTOK * NTOK * 4);

    // 1) prep: weight conversion + bias-table gather
    prep_kernel<<<(2 * DIMC * DIMC + 255) / 256, 256, 0, stream>>>(
        wq, wkv, wproj, rpi, bias_table, wq16, wkv16, wproj16, rpb);

    const int mtiles = (BWIN * NTOK) / 128;   // 2048

    // 2) q = x @ wq^T (scaled)
    gemm_wmma<0><<<dim3(DIMC / 64, mtiles), 256, 0, stream>>>(
        x, wq16, bq, q16, nullptr);

    // 3) kv = cross_x @ wkv^T  -> k16 / v16t
    gemm_wmma<1><<<dim3(2 * DIMC / 64, mtiles), 256, 0, stream>>>(
        cross_x, wkv16, bkv, k16, v16t);

    // 4) attention per (window, head)
    attn_kernel<<<BWIN * HEADS, 128, 0, stream>>>(
        q16, k16, v16t, rpb, mask, att16);

    // 5) out = att @ wproj^T + bproj (f32)
    gemm_wmma<2><<<dim3(DIMC / 64, mtiles), 256, 0, stream>>>(
        att16, wproj16, bproj, out, nullptr);
}
